// TemplatePairStackIteration_30270929502441
// MI455X (gfx1250) — compile-verified, matrix operand order checked by reading, not run on hardware
//
#include <hip/hip_runtime.h>
#include <hip/hip_bf16.h>
#include <cstdint>

typedef __bf16 bf16_t;
typedef __attribute__((ext_vector_type(16))) __bf16 v16bf;
typedef __attribute__((ext_vector_type(8)))  __bf16 v8bf;
typedef __attribute__((ext_vector_type(8)))  float  v8f;

#define NR   320
#define CF   64
#define NH   4
#define DC   32
#define HDC  128
#define MID  64
#define RTOT (NR*NR)

__device__ __forceinline__ float sigm(float x){ return 1.f/(1.f+__expf(-x)); }

__device__ __forceinline__ v8f wmma_bf16(v16bf a, v16bf b, v8f c){
  return __builtin_amdgcn_wmma_f32_16x16x32_bf16(false, a, false, b, (short)0, c, false, false);
}

// ---------------- async global->LDS copy (16B per lane), with fallback ----------------
#if __has_builtin(__builtin_amdgcn_global_load_async_to_lds_b128)
#define ASYNC_LDS 1
#else
#define ASYNC_LDS 0
#endif

typedef __attribute__((__vector_size__(4 * sizeof(int)))) int v4i_t;
typedef __attribute__((address_space(1))) v4i_t* gas_v4i;   // global (addrspace 1)
typedef __attribute__((address_space(3))) v4i_t* las_v4i;   // LDS (addrspace 3)

__device__ __forceinline__ void async_cp16(const bf16_t* g, bf16_t* l){
#if ASYNC_LDS
  __builtin_amdgcn_global_load_async_to_lds_b128(
      (gas_v4i)((v4i_t*)(void*)g),
      (las_v4i)((v4i_t*)(void*)l), 0, 0);
#else
  *(v8bf*)l = *(const v8bf*)g;     // global_load_b128 + ds_store_b128
#endif
}

template <int N>
__device__ __forceinline__ void wait_async(){
#if ASYNC_LDS
#if __has_builtin(__builtin_amdgcn_s_wait_asynccnt)
  __builtin_amdgcn_s_wait_asynccnt((short)N);
#else
  asm volatile("s_wait_asynccnt %0" :: "i"(N) : "memory");
#endif
#endif
}

// ---------------- LayerNorm: one wave per row of [rows, 64] ----------------
__global__ void k_layernorm(const float* __restrict__ x, const float* __restrict__ s,
                            const float* __restrict__ b, bf16_t* __restrict__ out, int rows){
  int wave = blockIdx.x * (blockDim.x >> 5) + (threadIdx.x >> 5);
  int lane = threadIdx.x & 31;
  if (wave >= rows) return;
  const float* xr = x + (size_t)wave * CF;
  float a0 = xr[lane], a1 = xr[lane + 32];
  float sum = a0 + a1;
  #pragma unroll
  for (int o = 16; o > 0; o >>= 1) sum += __shfl_xor(sum, o, 32);
  float mean = sum * (1.f / CF);
  float d0 = a0 - mean, d1 = a1 - mean;
  float vv = d0*d0 + d1*d1;
  #pragma unroll
  for (int o = 16; o > 0; o >>= 1) vv += __shfl_xor(vv, o, 32);
  float inv = rsqrtf(vv * (1.f / CF) + 1e-5f);
  bf16_t* orow = out + (size_t)wave * CF;
  orow[lane]      = (bf16_t)(d0 * inv * s[lane]      + b[lane]);
  orow[lane + 32] = (bf16_t)(d1 * inv * s[lane + 32] + b[lane + 32]);
}

// ---------------- Pack weight f32 [K,Ncol] -> bf16 WT [Ncol,K] ----------------
__global__ void k_pack_wt(const float* __restrict__ w, bf16_t* __restrict__ wt, int K, int Ncol){
  int i = blockIdx.x * blockDim.x + threadIdx.x;
  if (i >= K * Ncol) return;
  int k = i / Ncol, n = i % Ncol;
  wt[n * K + k] = (bf16_t)w[i];
}

// ---------------- Generic WMMA GEMM: out = A[R,K] * BT[Ncol,K]^T ----------------
// flags: 1 = add bias[Ncol]; 2 = sigmoid; 4 = bf16 output; 8 = add residual f32
__global__ void k_gemm(const bf16_t* __restrict__ A, const bf16_t* __restrict__ BT,
                       const float* __restrict__ bias, const float* __restrict__ resid,
                       float* __restrict__ outF, bf16_t* __restrict__ outB,
                       int R, int K, int Ncol, int flags){
  const int wid = threadIdx.x >> 5, lane = threadIdx.x & 31;
  const int row0 = (blockIdx.x * 4 + wid) * 16;
  const int col0 = blockIdx.y * 16;
  if (row0 >= R) return;
  const int ar = lane & 15, ks = (lane >> 4) * 16;
  const bf16_t* Ap = A  + (size_t)(row0 + ar) * K + ks;
  const bf16_t* Bp = BT + (size_t)(col0 + ar) * K + ks;
  v8f acc = {0,0,0,0,0,0,0,0};
  for (int k0 = 0; k0 < K; k0 += 32){
    v16bf af = *(const v16bf*)(Ap + k0);
    v16bf bf = *(const v16bf*)(Bp + k0);
    acc = wmma_bf16(af, bf, acc);
  }
  const int m0 = 8 * (lane >> 4), nn = lane & 15;
  const float bv = (flags & 1) ? bias[col0 + nn] : 0.f;
  #pragma unroll
  for (int r = 0; r < 8; ++r){
    float v = acc[r] + bv;
    if (flags & 2) v = sigm(v);
    size_t idx = (size_t)(row0 + m0 + r) * Ncol + col0 + nn;
    if (flags & 8) v += resid[idx];
    if (flags & 4) outB[idx] = (bf16_t)v;
    else           outF[idx] = v;
  }
}

// ---------------- Attention pair bias: bias[h][i*N+j] = xln . bw[:,h] ----------------
__global__ void k_bias_attn(const bf16_t* __restrict__ xln, const float* __restrict__ bw,
                            float* __restrict__ bias){
  size_t r = (size_t)blockIdx.x * blockDim.x + threadIdx.x;
  if (r >= (size_t)RTOT) return;
  float a0 = 0, a1 = 0, a2 = 0, a3 = 0;
  const bf16_t* xr = xln + r * CF;
  for (int c = 0; c < CF; ++c){
    float xv = (float)xr[c];
    a0 += xv * bw[c*NH + 0]; a1 += xv * bw[c*NH + 1];
    a2 += xv * bw[c*NH + 2]; a3 += xv * bw[c*NH + 3];
  }
  bias[0*(size_t)RTOT + r] = a0; bias[1*(size_t)RTOT + r] = a1;
  bias[2*(size_t)RTOT + r] = a2; bias[3*(size_t)RTOT + r] = a3;
}

// ---------------- v [m, j, hc] -> vt [m, hc, j] (bf16) ----------------
__global__ void k_transpose_v(const bf16_t* __restrict__ vb, bf16_t* __restrict__ vt){
  size_t id = (size_t)blockIdx.x * blockDim.x + threadIdx.x;
  if (id >= (size_t)RTOT * HDC) return;
  int j  = (int)(id % NR);
  int hc = (int)((id / NR) % HDC);
  int m  = (int)(id / ((size_t)NR * HDC));
  vt[id] = vb[((size_t)m * NR + j) * HDC + hc];
}

// ---------------- Fused attention core: one wave per (i-tile, head, row m) ----------------
__global__ void k_attn(const bf16_t* __restrict__ qb, const bf16_t* __restrict__ kb,
                       const bf16_t* __restrict__ vt, const bf16_t* __restrict__ gb,
                       const float* __restrict__ bias, bf16_t* __restrict__ obuf){
  __shared__ __align__(32) float  slog[16][NR];
  __shared__ __align__(32) bf16_t swgt[16][NR];
  const int lane = threadIdx.x & 31;
  const int it = blockIdx.x, h = blockIdx.y, m = blockIdx.z;
  const int i0 = it * 16;
  const int ar = lane & 15, ks = (lane >> 4) * 16;
  const int m0 = 8 * (lane >> 4), nn = lane & 15;

  // logits = q . k^T (single K=32 WMMA per 16x16 j-tile)
  v16bf qf = *(const v16bf*)(qb + ((size_t)m * NR + i0 + ar) * HDC + h * DC + ks);
  for (int jt = 0; jt < NR / 16; ++jt){
    v16bf kf = *(const v16bf*)(kb + ((size_t)m * NR + jt * 16 + ar) * HDC + h * DC + ks);
    v8f acc = {0,0,0,0,0,0,0,0};
    acc = wmma_bf16(qf, kf, acc);
    #pragma unroll
    for (int r = 0; r < 8; ++r) slog[m0 + r][jt * 16 + nn] = acc[r];
  }
  __syncthreads();

  // softmax over j (scale + pair bias), two lanes per row
  const int row = lane & 15, half = lane >> 4;
  const int jb = half * (NR / 2), je = jb + NR / 2;
  const float scale = 0.1767766952966369f;   // 1/sqrt(DC)
  const float* brow = bias + (size_t)h * RTOT + (size_t)(i0 + row) * NR;
  float mx = -1e30f;
  for (int j = jb; j < je; ++j){
    float v = slog[row][j] * scale + brow[j];
    slog[row][j] = v;
    mx = fmaxf(mx, v);
  }
  mx = fmaxf(mx, __shfl_xor(mx, 16, 32));
  float sum = 0.f;
  for (int j = jb; j < je; ++j){
    float e = __expf(slog[row][j] - mx);
    slog[row][j] = e;
    sum += e;
  }
  sum += __shfl_xor(sum, 16, 32);
  float inv = 1.f / sum;
  for (int j = jb; j < je; ++j) swgt[row][j] = (bf16_t)(slog[row][j] * inv);
  __syncthreads();

  // o = W . V  (A-fragments from LDS), fused sigmoid gate
  for (int ct = 0; ct < 2; ++ct){
    v8f acc = {0,0,0,0,0,0,0,0};
    for (int k0 = 0; k0 < NR; k0 += 32){
      v16bf wf = *(const v16bf*)(&swgt[ar][k0 + ks]);
      v16bf vf = *(const v16bf*)(vt + ((size_t)m * HDC + h * DC + ct * 16 + ar) * NR + k0 + ks);
      acc = wmma_bf16(wf, vf, acc);
    }
    #pragma unroll
    for (int r = 0; r < 8; ++r){
      size_t ridx = (size_t)m * NR + i0 + m0 + r;
      float gv = sigm((float)gb[ridx * HDC + h * DC + ct * 16 + nn]);
      obuf[ridx * HDC + h * DC + ct * 16 + nn] = (bf16_t)(acc[r] * gv);
    }
  }
}

// ---------------- f32 [i,j,c] -> [j,i,c] transpose ----------------
__global__ void k_transpose_x(const float* __restrict__ src, float* __restrict__ dst){
  size_t id = (size_t)blockIdx.x * blockDim.x + threadIdx.x;
  if (id >= (size_t)RTOT * CF) return;
  int c = (int)(id % CF);
  size_t r = id / CF;
  int j = (int)(r % NR), i = (int)(r / NR);
  dst[((size_t)j * NR + i) * CF + c] = src[id];
}

// ---------------- GLU + channel-major pack: p[c][i*N+k] = a*sigmoid(b) ----------------
__global__ void k_glu_pack(const float* __restrict__ a, const float* __restrict__ b,
                           bf16_t* __restrict__ p, int trans){
  size_t id = (size_t)blockIdx.x * blockDim.x + threadIdx.x;
  if (id >= (size_t)RTOT * MID) return;
  int c = (int)(id % MID);
  size_t r = id / MID;
  int k = (int)(r % NR), i = (int)(r / NR);
  float v = a[id] * sigm(b[id]);
  size_t dst = (size_t)c * RTOT + (trans ? ((size_t)k * NR + i) : r);
  p[dst] = (bf16_t)v;
}

// ---------------- Triangle einsum: o[i,j,c] = sum_k A_c[i,k] * B_c[j,k] ----------------
// Block: 128 threads (4 waves). 64x64 output tile per channel.
// A/B chunks (64 rows x 64 k) staged in LDS via async global->LDS, double-buffered.
#define TK   64
#define LSTR 72   // padded LDS row stride (bf16 elems) to skew banks

__global__ void k_tri_gemm(const bf16_t* __restrict__ pa, const bf16_t* __restrict__ pb,
                           float* __restrict__ o){
  __shared__ __align__(16) bf16_t sA[2][64 * LSTR];
  __shared__ __align__(16) bf16_t sB[2][64 * LSTR];
  const int tid = threadIdx.x;
  const int wid = tid >> 5, lane = tid & 31;
  const int ib0 = blockIdx.x * 64;
  const int jb0 = blockIdx.y * 64;
  const int c   = blockIdx.z;
  const bf16_t* Ag = pa + (size_t)c * RTOT + (size_t)ib0 * NR;
  const bf16_t* Bg = pb + (size_t)c * RTOT + (size_t)jb0 * NR;

  // copy mapping: 128 threads; each round covers 16 rows x 8 segments of 16B
  const int seg = tid & 7;        // 16B segment within 64-elem k-chunk
  const int rr  = tid >> 3;       // 0..15

  auto issue = [&](int kc, int buf){
    const int k0 = kc * TK;
    #pragma unroll
    for (int r4 = 0; r4 < 4; ++r4){
      const int row = r4 * 16 + rr;
      async_cp16(Ag + (size_t)row * NR + k0 + seg * 8, &sA[buf][row * LSTR + seg * 8]);
      async_cp16(Bg + (size_t)row * NR + k0 + seg * 8, &sB[buf][row * LSTR + seg * 8]);
    }
  };

  issue(0, 0);
  issue(1, 1);

  const int ar = lane & 15, ks = (lane >> 4) * 16;
  v8f zero = {0,0,0,0,0,0,0,0};
  v8f acc[4] = {zero, zero, zero, zero};

  #pragma unroll
  for (int kc = 0; kc < NR / TK; ++kc){           // 5 chunks
    if (kc < NR / TK - 1) wait_async<8>();        // chunk kc landed, kc+1 may fly
    else                  wait_async<0>();
    __syncthreads();
    const int buf = kc & 1;
    const bf16_t* la = &sA[buf][(wid * 16 + ar) * LSTR + ks];
    #pragma unroll
    for (int kk = 0; kk < 2; ++kk){               // two K=32 steps per chunk
      v16bf af = *(const v16bf*)(la + kk * 32);
      #pragma unroll
      for (int jt = 0; jt < 4; ++jt){
        v16bf bf = *(const v16bf*)(&sB[buf][(jt * 16 + ar) * LSTR + ks + kk * 32]);
        acc[jt] = wmma_bf16(af, bf, acc[jt]);
      }
    }
    __syncthreads();                              // all waves done reading buf
    if (kc + 2 < NR / TK) issue(kc + 2, buf);
  }

  const int m0 = 8 * (lane >> 4), nn = lane & 15;
  #pragma unroll
  for (int jt = 0; jt < 4; ++jt){
    #pragma unroll
    for (int r = 0; r < 8; ++r)
      o[((size_t)(ib0 + wid * 16 + m0 + r) * NR + jb0 + jt * 16 + nn) * MID + c] = acc[jt][r];
  }
}

// ---------------- x += proj * gate (gate already sigmoided) ----------------
__global__ void k_gate_add(const float* __restrict__ a, const float* __restrict__ g,
                           float* __restrict__ x, long n){
  long id = (long)blockIdx.x * blockDim.x + threadIdx.x;
  if (id >= n) return;
  x[id] += a[id] * g[id];
}

// =====================================================================================
extern "C" void kernel_launch(void* const* d_in, const int* in_sizes, int n_in,
                              void* d_out, int out_size, void* d_ws, size_t ws_size,
                              hipStream_t stream){
  (void)in_sizes; (void)n_in; (void)out_size; (void)ws_size;
  const float* x2d = (const float*)d_in[0];
  auto F = [&](int i){ return (const float*)d_in[i]; };

  // JAX pytree flattening: dict keys sorted. Top level: end(1..10), ingoing(11..26),
  // outgoing(27..42), start(43..52).
  struct AttnP { const float *bw,*gb,*gw,*kw,*nb,*ns,*ob,*ow,*qw,*vw; };
  struct MultP { const float *ib,*isb,*isw,*iw,*jb,*jsb,*jsw,*jw,*n1b,*n1s,*n2b,*n2s,*pb,*pw,*sb,*sw; };
  auto attnAt = [&](int b){ AttnP p{F(b+0),F(b+1),F(b+2),F(b+3),F(b+4),F(b+5),F(b+6),F(b+7),F(b+8),F(b+9)}; return p; };
  auto multAt = [&](int b){ MultP p{F(b+0),F(b+1),F(b+2),F(b+3),F(b+4),F(b+5),F(b+6),F(b+7),
                                    F(b+8),F(b+9),F(b+10),F(b+11),F(b+12),F(b+13),F(b+14),F(b+15)}; return p; };
  AttnP Pend = attnAt(1), Pstart = attnAt(43);
  MultP Ping = multAt(11), Pout = multAt(27);

  // workspace carve-up
  char* ws = (char*)d_ws;
  size_t off = 0;
  auto alloc = [&](size_t bytes){ size_t r = off; off = (off + bytes + 255) & ~(size_t)255; return r; };
  bf16_t* XLN   = (bf16_t*)(ws + alloc((size_t)RTOT * CF  * 2));
  bf16_t* QB    = (bf16_t*)(ws + alloc((size_t)RTOT * HDC * 2));
  bf16_t* KB    = (bf16_t*)(ws + alloc((size_t)RTOT * HDC * 2));
  bf16_t* VB    = (bf16_t*)(ws + alloc((size_t)RTOT * HDC * 2));
  bf16_t* GB    = (bf16_t*)(ws + alloc((size_t)RTOT * HDC * 2));
  bf16_t* VT    = (bf16_t*)(ws + alloc((size_t)RTOT * HDC * 2));
  bf16_t* OBUF  = (bf16_t*)(ws + alloc((size_t)RTOT * HDC * 2));
  float*  BIAS  = (float* )(ws + alloc((size_t)NH * RTOT * 4));
  float*  XT    = (float* )(ws + alloc((size_t)RTOT * CF * 4));
  float*  TMPA  = (float* )(ws + alloc((size_t)RTOT * CF * 4));
  float*  TMPB  = (float* )(ws + alloc((size_t)RTOT * CF * 4));
  bf16_t* PA    = (bf16_t*)(ws + alloc((size_t)MID * RTOT * 2));
  bf16_t* PB    = (bf16_t*)(ws + alloc((size_t)MID * RTOT * 2));
  float*  TRIO  = (float* )(ws + alloc((size_t)RTOT * MID * 4));
  bf16_t* TRLN  = (bf16_t*)(ws + alloc((size_t)RTOT * MID * 2));
  bf16_t* WP[6];
  for (int i = 0; i < 6; ++i) WP[i] = (bf16_t*)(ws + alloc(16384));

  const unsigned gLN   = (RTOT + 7) / 8;
  const unsigned gEl   = (unsigned)(((size_t)RTOT * HDC + 255) / 256);
  const unsigned gEl64 = (unsigned)(((size_t)RTOT * CF  + 255) / 256);

  auto runAttn = [&](const float* xin, float* xout, const AttnP& P){
    k_layernorm<<<gLN, 256, 0, stream>>>(xin, P.ns, P.nb, XLN, RTOT);
    k_pack_wt<<<(CF*HDC + 255)/256, 256, 0, stream>>>(P.qw, WP[0], CF, HDC);
    k_pack_wt<<<(CF*HDC + 255)/256, 256, 0, stream>>>(P.kw, WP[1], CF, HDC);
    k_pack_wt<<<(CF*HDC + 255)/256, 256, 0, stream>>>(P.vw, WP[2], CF, HDC);
    k_pack_wt<<<(CF*HDC + 255)/256, 256, 0, stream>>>(P.gw, WP[3], CF, HDC);
    k_pack_wt<<<(HDC*CF + 255)/256, 256, 0, stream>>>(P.ow, WP[4], HDC, CF);
    dim3 g128(RTOT/64, HDC/16);
    k_gemm<<<g128, 128, 0, stream>>>(XLN, WP[0], nullptr, nullptr, nullptr, QB, RTOT, CF, HDC, 4);
    k_gemm<<<g128, 128, 0, stream>>>(XLN, WP[1], nullptr, nullptr, nullptr, KB, RTOT, CF, HDC, 4);
    k_gemm<<<g128, 128, 0, stream>>>(XLN, WP[2], nullptr, nullptr, nullptr, VB, RTOT, CF, HDC, 4);
    k_gemm<<<g128, 128, 0, stream>>>(XLN, WP[3], P.gb,    nullptr, nullptr, GB, RTOT, CF, HDC, 4|1);
    k_bias_attn<<<(RTOT + 255)/256, 256, 0, stream>>>(XLN, P.bw, BIAS);
    k_transpose_v<<<gEl, 256, 0, stream>>>(VB, VT);
    k_attn<<<dim3(NR/16, NH, NR), 32, 0, stream>>>(QB, KB, VT, GB, BIAS, OBUF);
    dim3 g64(RTOT/64, CF/16);
    k_gemm<<<g64, 128, 0, stream>>>(OBUF, WP[4], P.ob, xin, xout, nullptr, RTOT, HDC, CF, 1|8);
  };

  auto runMult = [&](float* x, const MultP& P, bool outgoing){
    k_layernorm<<<gLN, 256, 0, stream>>>(x, P.n1s, P.n1b, XLN, RTOT);
    k_pack_wt<<<(CF*MID + 255)/256, 256, 0, stream>>>(P.iw,  WP[0], CF, MID);
    k_pack_wt<<<(CF*MID + 255)/256, 256, 0, stream>>>(P.isw, WP[1], CF, MID);
    k_pack_wt<<<(CF*MID + 255)/256, 256, 0, stream>>>(P.jw,  WP[2], CF, MID);
    k_pack_wt<<<(CF*MID + 255)/256, 256, 0, stream>>>(P.jsw, WP[3], CF, MID);
    k_pack_wt<<<(MID*CF + 255)/256, 256, 0, stream>>>(P.pw,  WP[4], MID, CF);
    k_pack_wt<<<(CF*CF  + 255)/256, 256, 0, stream>>>(P.sw,  WP[5], CF, CF);
    dim3 g64(RTOT/64, MID/16);
    int tflag = outgoing ? 0 : 1;
    // I = (xln@iw+ib) * sigmoid(xln@isw+isb)
    k_gemm<<<g64, 128, 0, stream>>>(XLN, WP[0], P.ib,  nullptr, TMPA, nullptr, RTOT, CF, MID, 1);
    k_gemm<<<g64, 128, 0, stream>>>(XLN, WP[1], P.isb, nullptr, TMPB, nullptr, RTOT, CF, MID, 1);
    k_glu_pack<<<gEl64, 256, 0, stream>>>(TMPA, TMPB, outgoing ? PA : PB, tflag);
    // J = (xln@jw+jb) * sigmoid(xln@jsw+jsb)
    k_gemm<<<g64, 128, 0, stream>>>(XLN, WP[2], P.jb,  nullptr, TMPA, nullptr, RTOT, CF, MID, 1);
    k_gemm<<<g64, 128, 0, stream>>>(XLN, WP[3], P.jsb, nullptr, TMPB, nullptr, RTOT, CF, MID, 1);
    k_glu_pack<<<gEl64, 256, 0, stream>>>(TMPA, TMPB, outgoing ? PB : PA, tflag);
    // o[i,j,c] = sum_k A_c[i,k] * B_c[j,k]  (async-LDS staged, double-buffered)
    k_tri_gemm<<<dim3(NR/64, NR/64, MID), 128, 0, stream>>>(PA, PB, TRIO);
    // LN over c, project, gate, residual add
    k_layernorm<<<gLN, 256, 0, stream>>>(TRIO, P.n2s, P.n2b, TRLN, RTOT);
    k_gemm<<<g64, 128, 0, stream>>>(TRLN, WP[4], P.pb, nullptr, TMPA, nullptr, RTOT, MID, CF, 1);
    k_gemm<<<g64, 128, 0, stream>>>(XLN,  WP[5], P.sb, nullptr, TMPB, nullptr, RTOT, CF,  CF, 1|2);
    k_gate_add<<<gEl64, 256, 0, stream>>>(TMPA, TMPB, x, (long)RTOT * CF);
  };

  float* X = (float*)d_out;
  // x = x + tri_attn(x, start)
  runAttn(x2d, X, Pstart);
  // x = x + swap(tri_attn(swap(x), end))
  k_transpose_x<<<gEl64, 256, 0, stream>>>(X, XT);
  runAttn(XT, XT, Pend);
  k_transpose_x<<<gEl64, 256, 0, stream>>>(XT, X);
  // x = x + tri_mult(x, outgoing); x = x + tri_mult(x, ingoing)
  runMult(X, Pout, true);
  runMult(X, Ping, false);
}
